// ELMo_1365799600600
// MI455X (gfx1250) — matrix operand order repaired and checked by reading, not compiled, not run on hardware
//
#include <hip/hip_runtime.h>
#include <hip/hip_fp16.h>

typedef __attribute__((ext_vector_type(16))) _Float16 v16h;
typedef __attribute__((ext_vector_type(8)))  _Float16 v8h;
typedef __attribute__((ext_vector_type(8)))  float    v8f;
typedef __attribute__((ext_vector_type(2)))  float    v2f;
typedef __attribute__((ext_vector_type(4)))  int      v4i;

#define BATCH 64
#define TFULL 256
#define TM    255
#define UU    256
#define G4    1024
#define SLOT  (BATCH*TM*UU)   // 4177920 floats per output slot

#define AS1 __attribute__((address_space(1)))
#define AS3 __attribute__((address_space(3)))

#if defined(__has_builtin)
#if __has_builtin(__builtin_amdgcn_global_load_async_to_lds_b128)
#define HAVE_ASYNC_LDS 1
#endif
#endif

#ifdef HAVE_ASYNC_LDS
__device__ __forceinline__ void wait_async0() {
#if __has_builtin(__builtin_amdgcn_s_wait_asynccnt)
    __builtin_amdgcn_s_wait_asynccnt(0);
#else
    asm volatile("s_wait_asynccnt 0x0" ::: "memory");
#endif
}
__device__ __forceinline__ void async_b128(const void* g, void* l) {
    __builtin_amdgcn_global_load_async_to_lds_b128(
        (AS1 v4i*)g, (AS3 v4i*)l, 0, 0);
}
#endif

struct F2 { float x, y; };

// ---------------- init: zero sync counters ----------------
__global__ void k_init(unsigned* sync) {
    if (threadIdx.x < 16) sync[threadIdx.x] = 0u;
}

// ---------------- embedding gather -> slots 0 and 3 ----------------
__global__ void k_embed(const int* __restrict__ seqs, const float* __restrict__ E,
                        float* __restrict__ out) {
    int blk = blockIdx.x;            // b*256 + t
    int b = blk >> 8, t = blk & 255;
    int u = threadIdx.x;
    int tok = seqs[blk];
    float v = E[(long)tok * UU + u];
    if (t < TM) out[0L*SLOT + ((long)b*TM + t)*UU + u] = v;       // fxs[0] = emb[:, :-1]
    if (t >= 1) out[3L*SLOT + ((long)b*TM + (t-1))*UU + u] = v;   // bxs[0] = emb[:, 1:]
}

// ---------------- transpose + f16 convert of W (both dirs) ----------------
__global__ void k_wt(const float* __restrict__ Wf, const float* __restrict__ Wb,
                     _Float16* __restrict__ WtF, _Float16* __restrict__ WtB) {
    int n = blockIdx.x;      // 0..1023
    int k = threadIdx.x;     // 0..255
    WtF[n*UU + k] = (_Float16)Wf[(long)k*G4 + n];
    WtB[n*UU + k] = (_Float16)Wb[(long)k*G4 + n];
}

// ---------------- zx = x @ W + b  (f16 WMMA, f32 accumulate) ----------------
// grid = 255*8 blocks, 512 threads. Block tile: 64 rows (all batches of one t) x 128 cols.
// zx stored t-major: row = t*64 + b, as f16.
__global__ __launch_bounds__(512)
void k_zx(const float* __restrict__ x, const _Float16* __restrict__ Wt,
          const float* __restrict__ bias, _Float16* __restrict__ zx) {
    __shared__ _Float16 As[64][40];    // 64 rows x 32 k, padded stride 40
    __shared__ _Float16 Bs[128][40];   // 128 cols x 32 k

    const int mb = blockIdx.x >> 3;    // t: 0..254
    const int nb = blockIdx.x & 7;     // N block: 0..7
    const int tid = threadIdx.x;
    const int lane = tid & 31, wave = tid >> 5;
    const int tm = wave >> 2;              // 0..3
    const int tn0 = (wave & 3) * 2;        // 0,2,4,6 (each wave does 2 N tiles)
    const int half = lane >> 4, l16 = lane & 15;

    v8f acc0 = {}; v8f acc1 = {};
    for (int kc = 0; kc < UU; kc += 32) {
        // stage A: x[b][t=mb][kc..kc+31] -> f16 (needs conversion, regular path)
        {
            int idx = tid * 4;
            int r = idx >> 5, kk = idx & 31;
            const float4 f = *(const float4*)&x[((long)r*TM + mb)*UU + kc + kk];
            As[r][kk+0] = (_Float16)f.x; As[r][kk+1] = (_Float16)f.y;
            As[r][kk+2] = (_Float16)f.z; As[r][kk+3] = (_Float16)f.w;
        }
        // stage B: Wt[nb*128 + c][kc..kc+31]  (pure f16 copy -> async to LDS)
        {
            int idx = tid * 8;
            int c = idx >> 5, kk = idx & 31;
#ifdef HAVE_ASYNC_LDS
            async_b128(&Wt[(long)(nb*128 + c)*UU + kc + kk], &Bs[c][kk]);
#else
            *(uint4*)&Bs[c][kk] = *(const uint4*)&Wt[(long)(nb*128 + c)*UU + kc + kk];
#endif
        }
#ifdef HAVE_ASYNC_LDS
        wait_async0();
#endif
        __syncthreads();

        // A fragment: K = chunk*16 + half*8 + e  (ISA 16-bit A layout)
        int m = tm*16 + l16;
        v8h a0 = *(const v8h*)&As[m][half*8];
        v8h a1 = *(const v8h*)&As[m][16 + half*8];
        v16h av = __builtin_shufflevector(a0, a1, 0,1,2,3,4,5,6,7,8,9,10,11,12,13,14,15);
        // B fragment: K = chunk*16 + half*8 + e  (same layout as A)
        int n0c = tn0*16 + l16;
        v8h b0a = *(const v8h*)&Bs[n0c][half*8];
        v8h b0b = *(const v8h*)&Bs[n0c][16 + half*8];
        v16h bv0 = __builtin_shufflevector(b0a, b0b, 0,1,2,3,4,5,6,7,8,9,10,11,12,13,14,15);
        int n1c = n0c + 16;
        v8h b1a = *(const v8h*)&Bs[n1c][half*8];
        v8h b1b = *(const v8h*)&Bs[n1c][16 + half*8];
        v16h bv1 = __builtin_shufflevector(b1a, b1b, 0,1,2,3,4,5,6,7,8,9,10,11,12,13,14,15);

        acc0 = __builtin_amdgcn_wmma_f32_16x16x32_f16(false, av, false, bv0, (short)0, acc0, false, false);
        acc1 = __builtin_amdgcn_wmma_f32_16x16x32_f16(false, av, false, bv1, (short)0, acc1, false, false);
        __syncthreads();
    }
    // store (t-major), add bias
    int ncol0 = nb*128 + tn0*16 + l16;
    float bv0 = bias[ncol0], bv1 = bias[ncol0 + 16];
    for (int r = 0; r < 8; ++r) {
        int bidx = tm*16 + half*8 + r;          // batch index (C layout: M = r + 8*half)
        long row = (long)mb*64 + bidx;
        zx[row*G4 + ncol0]      = (_Float16)(acc0[r] + bv0);
        zx[row*G4 + ncol0 + 16] = (_Float16)(acc1[r] + bv1);
    }
}

// ---------------- recurrent scan (fp32 WMMA 16x16x4, U in LDS) ----------------
#define HSTR 260   // hbuf row stride (floats), 16B aligned, bank-spread
#define USTR 130   // Upk row stride (float2)
#define ZSTR 132   // zbuf row stride (floats)

__device__ __forceinline__ float sigmoidf_(float x) { return 1.f / (1.f + __expf(-x)); }

__global__ __launch_bounds__(1024)
void k_scan(int layer, const _Float16* __restrict__ zxF, const _Float16* __restrict__ zxB,
            const float* __restrict__ Ufl, const float* __restrict__ Ubl,
            const int* __restrict__ seqs, float* __restrict__ out,
            float* __restrict__ gH, unsigned* __restrict__ sync) {
    __shared__ F2    Upk[128][USTR];     // packed (k,k+1) pairs x 128 local z-cols  (~133 KB)
    __shared__ float hbuf[64][HSTR];     // full h state                              (~67 KB)
    __shared__ float zbuf[64][ZSTR];     // z slice for gate math                     (~34 KB)
    __shared__ float cst[64][32];        // owned c state                             (8 KB)

    const int g   = blockIdx.x;
    const int dir = g >> 3;              // 0 = fwd, 1 = bwd
    const int wg  = g & 7;               // which 32-col slice of h we own
    const int cb  = wg * 32;
    const _Float16* zx = dir ? zxB : zxF;
    const float* U = dir ? Ubl : Ufl;
    float* gh = gH + (long)dir * (2*BATCH*UU);     // double-buffered h exchange
    unsigned* cnt = sync + dir*8;
    unsigned* gen = sync + dir*8 + 1;

    const int tid  = threadIdx.x;
    const int lane = tid & 31, wave = tid >> 5;
    const int tm = wave >> 3, tn = wave & 7;       // 4x8 tile grid (64 x 128)
    const int half = lane >> 4, l16 = lane & 15;

    // load U slice: local col c -> global col (c/32)*256 + cb + (c%32), packed K pairs
    for (int i = tid; i < 128*128; i += 1024) {
        int c = i & 127, k2 = i >> 7;
        int n = ((c >> 5) * 256) + cb + (c & 31);
        F2 p; p.x = U[(long)(2*k2)*G4 + n]; p.y = U[(long)(2*k2+1)*G4 + n];
        Upk[k2][c] = p;
    }
    for (int i = tid; i < 64*HSTR; i += 1024) ((float*)hbuf)[i] = 0.f;
    for (int i = tid; i < 64*32;  i += 1024) ((float*)cst)[i]  = 0.f;
    __syncthreads();

    const int out_slot = dir ? (4 + layer) : (1 + layer);
    float* outp = out + (long)out_slot * SLOT;

    const int m  = tm*16 + l16;          // A-fragment batch row
    const int nl = tn*16 + l16;          // local z column for C/D
    const int q  = nl >> 5, j = nl & 31;
    const long zn = (long)q*256 + cb + j;

    for (int s = 0; s < TM; ++s) {
        const int t = dir ? (TM - 1 - s) : s;

        // ---- z_part = h @ U_slice : 64 chained v_wmma_f32_16x16x4_f32 ----
        v8f acc = {};
        for (int kk = 0; kk < 64; ++kk) {
            v2f a = *(const v2f*)&hbuf[m][kk*4 + half*2];  // A: K = kk*4 + half*2 + e
            F2 bp = Upk[kk*2 + half][nl];                  // B: same K pair, column nl
            v2f b; b[0] = bp.x; b[1] = bp.y;
            acc = __builtin_amdgcn_wmma_f32_16x16x4_f32(false, a, false, b, (short)0, acc, false, false);
        }
        // add zx slice (t-major), publish to zbuf
        for (int r = 0; r < 8; ++r) {
            int bidx = tm*16 + half*8 + r;
            zbuf[bidx][nl] = acc[r] + (float)zx[((long)t*64 + bidx)*G4 + zn];
        }
        __syncthreads();

        // ---- gates + mask-carry for owned 32 columns ----
        for (int p = tid; p < 2048; p += 1024) {
            int bb_ = p >> 5, jj = p & 31;
            int jg = cb + jj;
            float zi = zbuf[bb_][jj];
            float zf = zbuf[bb_][32 + jj];
            float zg = zbuf[bb_][64 + jj];
            float zo = zbuf[bb_][96 + jj];
            float c_old = cst[bb_][jj];
            float h_old = hbuf[bb_][jg];
            float c_new = sigmoidf_(zf) * c_old + sigmoidf_(zi) * tanhf(zg);
            float h_new = sigmoidf_(zo) * tanhf(c_new);
            bool mt = seqs[bb_*TFULL + t + dir] != 0;   // fm[b][t] / bm[b][t]
            float h2 = mt ? h_new : h_old;
            cst[bb_][jj] = mt ? c_new : c_old;
            outp[((long)bb_*TM + t)*UU + jg] = h2;
            gh[(s & 1)*(BATCH*UU) + bb_*UU + jg] = h2;
        }

        // prefetch next step's zx slice into cache
        if (s + 1 < TM && tid < 256) {
            int tn2 = dir ? (TM - 2 - s) : (s + 1);
            int row = tid >> 2, qq = tid & 3;
            __builtin_prefetch(&zx[((long)tn2*64 + row)*G4 + qq*256 + cb], 0, 0);
        }

        // ---- grid barrier among the 8 WGs of this scan ----
        // release: every thread waits for its own stores to reach agent scope
        __threadfence();
        __syncthreads();
        if (tid == 0) {
            unsigned gv = __hip_atomic_load(gen, __ATOMIC_ACQUIRE, __HIP_MEMORY_SCOPE_AGENT);
            unsigned a  = __hip_atomic_fetch_add(cnt, 1u, __ATOMIC_ACQ_REL, __HIP_MEMORY_SCOPE_AGENT);
            if (a == 7u) {
                __hip_atomic_store(cnt, 0u, __ATOMIC_RELAXED, __HIP_MEMORY_SCOPE_AGENT);
                __hip_atomic_fetch_add(gen, 1u, __ATOMIC_ACQ_REL, __HIP_MEMORY_SCOPE_AGENT);
            } else {
                while (__hip_atomic_load(gen, __ATOMIC_ACQUIRE, __HIP_MEMORY_SCOPE_AGENT) == gv)
                    __builtin_amdgcn_s_sleep(1);
            }
        }
        __syncthreads();
        // acquire: every wave invalidates so fresh h is observed
        __builtin_amdgcn_fence(__ATOMIC_ACQUIRE, "agent");

        // ---- reload full h into LDS ----
        {
            const float* src = gh + (s & 1)*(BATCH*UU);
#ifdef HAVE_ASYNC_LDS
            for (int i = tid; i < (BATCH*UU)/4; i += 1024) {
                int q4 = i*4; int bb_ = q4 >> 8, col = q4 & 255;
                async_b128(&src[q4], &hbuf[bb_][col]);
            }
            wait_async0();
#else
            for (int i = tid; i < (BATCH*UU)/4; i += 1024) {
                float4 v = *(const float4*)&src[i*4];
                int q4 = i*4; int bb_ = q4 >> 8, col = q4 & 255;
                *(float4*)&hbuf[bb_][col] = v;
            }
#endif
        }
        __syncthreads();
    }
}

// ---------------- host-side orchestration ----------------
extern "C" void kernel_launch(void* const* d_in, const int* in_sizes, int n_in,
                              void* d_out, int out_size, void* d_ws, size_t ws_size,
                              hipStream_t stream) {
    const int*   seqs = (const int*)  d_in[0];
    const float* E    = (const float*)d_in[1];
    const float* Wf   = (const float*)d_in[2];
    const float* Uf   = (const float*)d_in[3];
    const float* bf   = (const float*)d_in[4];
    const float* Wb   = (const float*)d_in[5];
    const float* Ub   = (const float*)d_in[6];
    const float* bb   = (const float*)d_in[7];
    float* out = (float*)d_out;

    char* ws = (char*)d_ws;
    _Float16* zxF = (_Float16*)ws;  ws += (size_t)TM*BATCH*G4*2;   // 33.4 MB
    _Float16* zxB = (_Float16*)ws;  ws += (size_t)TM*BATCH*G4*2;   // 33.4 MB
    _Float16* WtF = (_Float16*)ws;  ws += (size_t)G4*UU*2;
    _Float16* WtB = (_Float16*)ws;  ws += (size_t)G4*UU*2;
    float*    gH  = (float*)ws;     ws += (size_t)4*BATCH*UU*4;    // 2 dirs x 2 bufs
    unsigned* syn = (unsigned*)ws;

    k_init<<<1, 64, 0, stream>>>(syn);
    k_embed<<<BATCH*TFULL, UU, 0, stream>>>(seqs, E, out);

    for (int l = 0; l < 2; ++l) {
        const size_t wo = (size_t)l * UU * G4;
        k_wt<<<G4, UU, 0, stream>>>(Wf + wo, Wb + wo, WtF, WtB);
        k_zx<<<TM*8, 512, 0, stream>>>(out + (size_t)l*SLOT,     WtF, bf + l*G4, zxF);
        k_zx<<<TM*8, 512, 0, stream>>>(out + (size_t)(3+l)*SLOT, WtB, bb + l*G4, zxB);
        k_scan<<<16, 1024, 0, stream>>>(l, zxF, zxB, Uf + wo, Ub + wo, seqs, out, gH, syn);
    }
}
